// LIDDetector_23012434772453
// MI455X (gfx1250) — matrix operand order repaired and checked by reading, compiled, not run on hardware
//
#include <hip/hip_runtime.h>
#include <stdint.h>

// ---------- CDNA5 vector types ----------
typedef __attribute__((ext_vector_type(16))) __bf16        v16bf;
typedef __attribute__((ext_vector_type(8)))  float         v8f;
typedef __attribute__((ext_vector_type(4)))  unsigned int  v4u;

#define NLAYERS 4
#define MREF    2048          // est rows per layer
#define BROWS   32768         // x rows
#define TILE_M  16            // x rows per workgroup
#define THREADS 256           // 8 waves (wave32)
#define DMAX    1024

// est layer element offset inside hi/lo workspace arrays:
// off(j) = MREF * 128 * ((1<<j)-1) = 262144*((1<<j)-1)
__device__ __forceinline__ size_t est_off(int j) {
    return (size_t)262144u * (size_t)((1u << j) - 1u);
}

// round-to-nearest-even f32 -> bf16 bits
__device__ __forceinline__ unsigned short f2bf(float f) {
    unsigned u = __float_as_uint(f);
    unsigned r = u + 0x7FFFu + ((u >> 16) & 1u);
    return (unsigned short)(r >> 16);
}
__device__ __forceinline__ float bf2f(unsigned short h) {
    return __uint_as_float((unsigned)h << 16);
}

// WGP-scope prefetch (SCOPE field = 0 by default): pulls into ALL cache
// levels, unlike __builtin_prefetch which lowers to SE/SYS scope (GL2-only).
__device__ __forceinline__ void prefetch_wgp64(const void* p) {
    asm volatile("global_prefetch_b8 %0, off offset:64" :: "v"(p));
}

// ---------------------------------------------------------------------------
// Prep: est -> bf16 (hi, lo) split + exact f32 row norms ||e||^2.
// One block per (layer, est-row).  grid = 4*2048, block = 128.
// ---------------------------------------------------------------------------
__global__ void lid_prep_est(const float* __restrict__ e0, const float* __restrict__ e1,
                             const float* __restrict__ e2, const float* __restrict__ e3,
                             unsigned short* __restrict__ ehi, unsigned short* __restrict__ elo,
                             float* __restrict__ ee)
{
    __shared__ float red[4];
    int bid = blockIdx.x;
    int j   = bid >> 11;
    int row = bid & (MREF - 1);
    int D   = 128 << j;
    const float* srcs[4] = { e0, e1, e2, e3 };
    const float* src = srcs[j] + (size_t)row * D;
    size_t off = est_off(j) + (size_t)row * D;

    int tid = threadIdx.x, lane = tid & 31, w = tid >> 5;
    float ss = 0.f;
    for (int c = tid; c < D; c += 128) {
        float f  = src[c];
        unsigned short h = f2bf(f);
        float r  = f - bf2f(h);
        ehi[off + c] = h;
        elo[off + c] = f2bf(r);
        ss = fmaf(f, f, ss);
    }
    #pragma unroll
    for (int o = 16; o; o >>= 1) ss += __shfl_xor(ss, o, 32);
    if (lane == 0) red[w] = ss;
    __syncthreads();
    if (tid == 0) ee[j * MREF + row] = red[0] + red[1] + red[2] + red[3];
}

// ---------------------------------------------------------------------------
// Main: per 16-row x-tile per layer:
//   (1) stage x tile as bf16 hi/lo in LDS, load ||e||^2, compute ||x||^2
//   (2) split-bf16 WMMA GEMM: s[n] = ||e_n||^2 - 2 * <x_m, e_n>  -> LDS
//   (3) fused top-(k+1) selection + LID score
// grid = 4 * (32768/16) = 8192, block = 256, dynSmem ~200 KB.
// ---------------------------------------------------------------------------
__global__ __launch_bounds__(THREADS, 1) void lid_main(
    const float* __restrict__ x0, const float* __restrict__ x1,
    const float* __restrict__ x2, const float* __restrict__ x3,
    const unsigned short* __restrict__ ehi, const unsigned short* __restrict__ elo,
    const float* __restrict__ ee, const int* __restrict__ kptr,
    float* __restrict__ out)
{
    extern __shared__ char smem[];
    float*          sdist = (float*)smem;                                   // 16*2048 f32 = 128 KB
    unsigned short* sxh   = (unsigned short*)(smem + TILE_M * MREF * 4);    // 16*1024 bf16 = 32 KB
    unsigned short* sxl   = sxh + TILE_M * DMAX;                            // 32 KB
    float*          see   = (float*)(smem + TILE_M * MREF * 4 + 2 * TILE_M * DMAX * 2); // 8 KB
    float*          sxx   = see + MREF;                                     // 16 f32

    int bid = blockIdx.x;
    int j   = bid >> 11;           // layer
    int rt  = bid & 2047;          // row tile
    int D   = 128 << j;
    const float* xs4[4] = { x0, x1, x2, x3 };
    const float* xsrc = xs4[j] + (size_t)rt * TILE_M * D;
    size_t eoff = est_off(j);

    int tid = threadIdx.x, lane = tid & 31, w = tid >> 5;

    // ---- phase 0: staging ----
    for (int i = tid; i < TILE_M * D; i += THREADS) {
        float f = xsrc[i];
        unsigned short h = f2bf(f);
        sxh[i] = h;
        sxl[i] = f2bf(f - bf2f(h));
    }
    for (int i = tid; i < MREF; i += THREADS) see[i] = ee[j * MREF + i];
    // exact ||x||^2 per row (2 rows per wave)
    #pragma unroll
    for (int rr = 0; rr < 2; ++rr) {
        int m = w * 2 + rr;
        const float* xr = xsrc + (size_t)m * D;
        float s = 0.f;
        for (int c = lane; c < D; c += 32) { float f = xr[c]; s = fmaf(f, f, s); }
        #pragma unroll
        for (int o = 16; o; o >>= 1) s += __shfl_xor(s, o, 32);
        if (lane == 0) sxx[m] = s;
    }
    __syncthreads();

    // ---- phase 1: split-bf16 WMMA GEMM ----
    // A (16xK, bf16): lane L holds M=L&15; K chunk [16s,16s+16), s=L>>4
    // B (Kx16, bf16): lane L holds N=L&15; K chunk [16s,16s+16)
    int sel   = lane >> 4;
    int mrow  = lane & 15;
    int abase = mrow * D;

    for (int tb = 0; tb < 4; ++tb) {               // 4 groups of 4 N-tiles per wave
        v8f acc[4];
        size_t brow[4];
        #pragma unroll
        for (int t4 = 0; t4 < 4; ++t4) {
            v8f z = {};
            acc[t4] = z;
            int n = w * 256 + tb * 64 + t4 * 16 + mrow;
            brow[t4] = eoff + (size_t)n * D + (size_t)(sel * 16);
        }
        for (int k0 = 0; k0 < D; k0 += 32) {
            union { v4u q[2]; v16bf v; } Ah, Al;
            int a0 = abase + k0 + sel * 16;
            Ah.q[0] = *(const v4u*)&sxh[a0];
            Ah.q[1] = *(const v4u*)&sxh[a0 + 8];
            Al.q[0] = *(const v4u*)&sxl[a0];
            Al.q[1] = *(const v4u*)&sxl[a0 + 8];
            #pragma unroll
            for (int t4 = 0; t4 < 4; ++t4) {
                const unsigned short* ph = ehi + brow[t4] + k0;
                const unsigned short* pl = elo + brow[t4] + k0;
                // WGP-scope prefetch of next K-step's B lines (hi and lo)
                prefetch_wgp64(ph);
                prefetch_wgp64(pl);
                union { v4u q[2]; v16bf v; } Bh, Bl;
                Bh.q[0] = *(const v4u*)(ph);  Bh.q[1] = *(const v4u*)(ph + 8);
                Bl.q[0] = *(const v4u*)(pl);  Bl.q[1] = *(const v4u*)(pl + 8);
                acc[t4] = __builtin_amdgcn_wmma_f32_16x16x32_bf16(
                    false, Ah.v, false, Bh.v, (short)0, acc[t4], false, false);
                acc[t4] = __builtin_amdgcn_wmma_f32_16x16x32_bf16(
                    false, Ah.v, false, Bl.v, (short)0, acc[t4], false, false);
                acc[t4] = __builtin_amdgcn_wmma_f32_16x16x32_bf16(
                    false, Al.v, false, Bh.v, (short)0, acc[t4], false, false);
            }
        }
        // selection key: s = ||e_n||^2 - 2*dot  (||x||^2 is per-row constant)
        #pragma unroll
        for (int t4 = 0; t4 < 4; ++t4) {
            int n = w * 256 + tb * 64 + t4 * 16 + mrow;
            float een = see[n];
            #pragma unroll
            for (int r = 0; r < 8; ++r) {
                int mm = r + 8 * sel;                  // C/D layout: M = r + 8*(lane>=16)
                sdist[mm * MREF + n] = fmaf(-2.f, acc[t4][r], een);
            }
        }
    }
    __syncthreads();

    // ---- phase 2: fused top-(k+1) selection + LID score (2 rows per wave) ----
    int kk = kptr[0];
    if (kk > MREF - 1) kk = MREF - 1;
    if (kk < 1) kk = 1;
    #pragma unroll
    for (int rr = 0; rr < 2; ++rr) {
        int m = w * 2 + rr;
        float  xxm = sxx[m];
        float* row = sdist + m * MREF;
        float sumlog = 0.f, lastlog = 0.f;
        for (int it = 0; it <= kk; ++it) {
            float best = 3.0e38f; int bidx = lane;
            for (int i = lane; i < MREF; i += 32) {
                float v = row[i];
                if (v < best) { best = v; bidx = i; }
            }
            #pragma unroll
            for (int o = 16; o; o >>= 1) {
                float ov = __shfl_xor(best, o, 32);
                int   oi = __shfl_xor(bidx, o, 32);
                if (ov < best || (ov == best && oi < bidx)) { best = ov; bidx = oi; }
            }
            // invalidate the extracted minimum (same-wave DS ops are in order)
            if (lane == (bidx & 31)) row[bidx] = 3.0e38f;
            float d2 = fmaxf(xxm + best, 1e-12f);
            float lg = __logf(d2);
            if (it >= 1) { sumlog += lg; lastlog = lg; }
        }
        if (lane == 0) {
            // sum log(v_i/v_k) with v = sqrt(d2) -> 0.5*(sum log d2 - k*log d2_k)
            float denom = 0.5f * (sumlog - (float)kk * lastlog) + 1e-8f;
            out[(size_t)(rt * TILE_M + m) * NLAYERS + j] = -(float)kk / denom;
        }
    }
}

// ---------------------------------------------------------------------------
// d_in order (setup_inputs dict order): x0,est0,x1,est1,x2,est2,x3,est3,k
// d_ws layout: est_hi (2048*1920 bf16) | est_lo (same) | ee (4*2048 f32) ~15.8 MB
// ---------------------------------------------------------------------------
extern "C" void kernel_launch(void* const* d_in, const int* in_sizes, int n_in,
                              void* d_out, int out_size, void* d_ws, size_t ws_size,
                              hipStream_t stream) {
    const float* x0 = (const float*)d_in[0];
    const float* e0 = (const float*)d_in[1];
    const float* x1 = (const float*)d_in[2];
    const float* e1 = (const float*)d_in[3];
    const float* x2 = (const float*)d_in[4];
    const float* e2 = (const float*)d_in[5];
    const float* x3 = (const float*)d_in[6];
    const float* e3 = (const float*)d_in[7];
    const int* kptr = (const int*)d_in[8];

    unsigned short* ehi = (unsigned short*)d_ws;
    unsigned short* elo = ehi + (size_t)MREF * 1920;
    float*          ee  = (float*)(elo + (size_t)MREF * 1920);

    lid_prep_est<<<NLAYERS * MREF, 128, 0, stream>>>(e0, e1, e2, e3, ehi, elo, ee);

    size_t shmem = (size_t)TILE_M * MREF * 4        // sdist
                 + (size_t)2 * TILE_M * DMAX * 2    // x hi/lo bf16
                 + (size_t)MREF * 4                 // ||e||^2
                 + (size_t)TILE_M * 4;              // ||x||^2
    lid_main<<<NLAYERS * (BROWS / TILE_M), THREADS, shmem, stream>>>(
        x0, x1, x2, x3, ehi, elo, ee, kptr, (float*)d_out);
}
